// TransformerGuidenceSharedAttentionDecoderLayer_89928025244016
// MI455X (gfx1250) — compile-verified, matrix operand order checked by reading, not run on hardware
//
#include <hip/hip_runtime.h>
#include <stdint.h>

#define DEVI __device__ __forceinline__

typedef __attribute__((ext_vector_type(16))) __bf16 v16bf;
typedef __attribute__((ext_vector_type(8)))  float  v8f;

constexpr int NQ = 1024, NKK = 2048, BN = 4, DM = 256, HN = 8, HDn = 32, DFF = 1024;
constexpr int MQ = NQ * BN;   // 4096 token rows (q side)
constexpr int MK = NKK * BN;  // 8192 token rows (kv side)
constexpr float SCALE = 0.17677669529663687f;  // 32^-0.5

// ---------- helpers ----------
DEVI uint16_t f2bf(float f) {
  uint32_t u = __float_as_uint(f);
  uint32_t r = u + 0x7FFFu + ((u >> 16) & 1u);   // round-to-nearest-even
  return (uint16_t)(r >> 16);
}

union FragU { uint32_t u[8]; v16bf v; };

// Load one wave's A- or B-fragment for v_wmma_*_16x16x32_bf16 from a
// row-major [rows, pitch] bf16 matrix. For A: row = M index (lane&15).
// For B: row = N index (lane&15), since W is stored [N, K] row-major.
// K-pair layout per ISA: K(j,hi,t) = (j>>2)*16 + hi*8 + (j&3)*2 + t.
DEVI v16bf load_frag(const uint16_t* mat, int row, int pitch, int k0, int hi) {
  FragU f;
  const uint32_t* p = (const uint32_t*)(mat + (size_t)row * (size_t)pitch + k0 + hi * 8);
#pragma unroll
  for (int j = 0; j < 8; ++j) f.u[j] = p[(j >> 2) * 8 + (j & 3)];
  return f.v;
}

DEVI v8f wmma_bf16(v16bf a, v16bf b, v8f c) {
  return __builtin_amdgcn_wmma_f32_16x16x32_bf16(false, a, false, b, (short)0, c,
                                                 false, false);
}

DEVI float half_max(float a) {           // max over the 16 lanes of this half-wave
#pragma unroll
  for (int m = 1; m < 16; m <<= 1) a = fmaxf(a, __shfl_xor(a, m));
  return a;
}
DEVI float half_sum(float a) {
#pragma unroll
  for (int m = 1; m < 16; m <<= 1) a += __shfl_xor(a, m);
  return a;
}

// ---------- conversion kernels ----------
__global__ void k_f2bf(const float* __restrict__ in, uint16_t* __restrict__ out, int n) {
  int i = blockIdx.x * blockDim.x + threadIdx.x;
  if (i < n) out[i] = f2bf(in[i]);
}
__global__ void k_add_f2bf(const float* __restrict__ a, const float* __restrict__ b,
                           uint16_t* __restrict__ out, int n) {
  int i = blockIdx.x * blockDim.x + threadIdx.x;
  if (i < n) out[i] = f2bf(b ? a[i] + b[i] : a[i]);
}

// ---------- LayerNorm (D = 256, one block per row) ----------
__global__ void k_ln(const float* __restrict__ x, const float* __restrict__ g,
                     const float* __restrict__ bt, const float* __restrict__ pos,
                     uint16_t* __restrict__ out_pos, uint16_t* __restrict__ out_plain) {
  int row = blockIdx.x, t = threadIdx.x;
  float v = x[(size_t)row * DM + t];
  __shared__ float red[DM];
  red[t] = v;
  __syncthreads();
#pragma unroll
  for (int s = 128; s > 0; s >>= 1) {
    if (t < s) red[t] += red[t + s];
    __syncthreads();
  }
  float mean = red[0] * (1.0f / DM);
  __syncthreads();
  float d = v - mean;
  red[t] = d * d;
  __syncthreads();
#pragma unroll
  for (int s = 128; s > 0; s >>= 1) {
    if (t < s) red[t] += red[t + s];
    __syncthreads();
  }
  float var = red[0] * (1.0f / DM);
  float y = d * rsqrtf(var + 1e-5f) * g[t] + bt[t];
  if (out_plain) out_plain[(size_t)row * DM + t] = f2bf(y);
  if (out_pos) out_pos[(size_t)row * DM + t] = f2bf(y + pos[(size_t)row * DM + t]);
}

// ---------- generic WMMA GEMM:  C[M,N] = A[M,K] * W[N,K]^T + bias ----------
// 4 waves/block, each wave owns 16 rows x 64 cols (4 accumulators, A reused 4x).
// Software-pipelined: next k-step fragments are in flight during current WMMAs.
__global__ void k_gemm(const uint16_t* __restrict__ A, const uint16_t* __restrict__ W,
                       const float* __restrict__ bias, const float* __restrict__ resid,
                       float* __restrict__ outf, uint16_t* __restrict__ outb,
                       int M, int N, int K, int relu) {
  int lane = threadIdx.x & 31, wave = threadIdx.x >> 5;
  int lo = lane & 15, hi = lane >> 4;
  int m0 = (blockIdx.x * 4 + wave) * 16;
  int n0 = blockIdx.y * 64;
  v8f acc[4] = {v8f{}, v8f{}, v8f{}, v8f{}};

  v16bf a = load_frag(A, m0 + lo, K, 0, hi);
  v16bf b0 = load_frag(W, n0 + lo, K, 0, hi);
  v16bf b1 = load_frag(W, n0 + 16 + lo, K, 0, hi);
  v16bf b2 = load_frag(W, n0 + 32 + lo, K, 0, hi);
  v16bf b3 = load_frag(W, n0 + 48 + lo, K, 0, hi);

  for (int k0 = 32; k0 < K; k0 += 32) {
    // prefetch next k-step while current WMMAs run
    v16bf an = load_frag(A, m0 + lo, K, k0, hi);
    v16bf c0 = load_frag(W, n0 + lo, K, k0, hi);
    v16bf c1 = load_frag(W, n0 + 16 + lo, K, k0, hi);
    v16bf c2 = load_frag(W, n0 + 32 + lo, K, k0, hi);
    v16bf c3 = load_frag(W, n0 + 48 + lo, K, k0, hi);
    acc[0] = wmma_bf16(a, b0, acc[0]);
    acc[1] = wmma_bf16(a, b1, acc[1]);
    acc[2] = wmma_bf16(a, b2, acc[2]);
    acc[3] = wmma_bf16(a, b3, acc[3]);
    a = an; b0 = c0; b1 = c1; b2 = c2; b3 = c3;
  }
  acc[0] = wmma_bf16(a, b0, acc[0]);
  acc[1] = wmma_bf16(a, b1, acc[1]);
  acc[2] = wmma_bf16(a, b2, acc[2]);
  acc[3] = wmma_bf16(a, b3, acc[3]);

#pragma unroll
  for (int t = 0; t < 4; ++t) {
    int col = n0 + t * 16 + lo;
    float bv = bias[col];
#pragma unroll
    for (int r = 0; r < 8; ++r) {
      size_t idx = (size_t)(m0 + r + 8 * hi) * N + col;
      float v = acc[t][r] + bv;
      if (relu) v = fmaxf(v, 0.0f);
      if (resid) v += resid[idx];
      if (outf) outf[idx] = v;
      if (outb) outb[idx] = f2bf(v);
    }
  }
}

// ---------- flash self-attention (one wave per 16-query tile per (b,h)) ----------
__global__ void k_flash_self(const uint16_t* __restrict__ Qm, const uint16_t* __restrict__ Km,
                             const uint16_t* __restrict__ Vm, uint16_t* __restrict__ Om,
                             int nkeys) {
  int lane = threadIdx.x, lo = lane & 15, hi = lane >> 4;
  int qt = blockIdx.x, b = blockIdx.y / HN, h = blockIdx.y % HN;
  const int pitch = BN * DM;  // token-row stride
  const uint16_t* Qp = Qm + b * DM + h * HDn;
  const uint16_t* Kp = Km + b * DM + h * HDn;
  const uint16_t* Vp = Vm + b * DM + h * HDn;
  uint16_t* Op = Om + b * DM + h * HDn;

  __shared__ __align__(16) uint16_t P[16 * 32];
  __shared__ __align__(16) uint16_t VT[32 * 32];

  v16bf qf = load_frag(Qp, qt * 16 + lo, pitch, 0, hi);
  float m[8], Z[8];
  v8f o0 = v8f{}, o1 = v8f{};
#pragma unroll
  for (int r = 0; r < 8; ++r) { m[r] = -1e30f; Z[r] = 0.0f; }

  for (int kk0 = 0; kk0 < nkeys; kk0 += 32) {
    v16bf kf0 = load_frag(Kp, kk0 + lo, pitch, 0, hi);
    v16bf kf1 = load_frag(Kp, kk0 + 16 + lo, pitch, 0, hi);
    // stage V chunk transposed (VT[hd][kk]); loads overlap with S WMMAs below
    {
      const uint16_t* vrow = Vp + (size_t)(kk0 + lane) * pitch;
#pragma unroll
      for (int i = 0; i < 32; ++i) VT[i * 32 + lane] = vrow[i];
    }
    v8f s0 = wmma_bf16(qf, kf0, v8f{});
    v8f s1 = wmma_bf16(qf, kf1, v8f{});
    float al[8];
#pragma unroll
    for (int r = 0; r < 8; ++r) {
      float rm = half_max(fmaxf(s0[r], s1[r]) * SCALE);
      float mn = fmaxf(m[r], rm);
      al[r] = __expf(m[r] - mn);
      float p0 = __expf(s0[r] * SCALE - mn);
      float p1 = __expf(s1[r] * SCALE - mn);
      Z[r] = Z[r] * al[r] + half_sum(p0 + p1);
      m[r] = mn;
      P[(r + 8 * hi) * 32 + lo] = f2bf(p0);
      P[(r + 8 * hi) * 32 + 16 + lo] = f2bf(p1);
    }
    v16bf pf = load_frag(P, lo, 32, 0, hi);
    v16bf vb0 = load_frag(VT, lo, 32, 0, hi);
    v16bf vb1 = load_frag(VT, 16 + lo, 32, 0, hi);
#pragma unroll
    for (int r = 0; r < 8; ++r) { o0[r] *= al[r]; o1[r] *= al[r]; }
    o0 = wmma_bf16(pf, vb0, o0);
    o1 = wmma_bf16(pf, vb1, o1);
  }
#pragma unroll
  for (int r = 0; r < 8; ++r) {
    float inv = 1.0f / Z[r];
    size_t rowoff = (size_t)(qt * 16 + r + 8 * hi) * pitch;
    Op[rowoff + lo] = f2bf(o0[r] * inv);
    Op[rowoff + 16 + lo] = f2bf(o1[r] * inv);
  }
}

// ---------- guidance pass 1: exact text softmax stats (m, Z) ----------
__global__ void k_guid_pass1(const uint16_t* __restrict__ Qm, const uint16_t* __restrict__ Km,
                             float* __restrict__ mrow, float* __restrict__ zrow) {
  int lane = threadIdx.x, lo = lane & 15, hi = lane >> 4;
  int qt = blockIdx.x, b = blockIdx.y / HN, h = blockIdx.y % HN;
  const int pitch = BN * DM;
  const uint16_t* Qp = Qm + b * DM + h * HDn;
  const uint16_t* Kp = Km + b * DM + h * HDn;
  v16bf qf = load_frag(Qp, qt * 16 + lo, pitch, 0, hi);
  float m[8], Z[8];
#pragma unroll
  for (int r = 0; r < 8; ++r) { m[r] = -1e30f; Z[r] = 0.0f; }

  v16bf kf0 = load_frag(Kp, lo, pitch, 0, hi);
  v16bf kf1 = load_frag(Kp, 16 + lo, pitch, 0, hi);
  for (int kk0 = 0; kk0 < NKK; kk0 += 32) {
    v8f s0 = wmma_bf16(qf, kf0, v8f{});
    v8f s1 = wmma_bf16(qf, kf1, v8f{});
    if (kk0 + 32 < NKK) {   // prefetch next chunk during softmax VALU work
      kf0 = load_frag(Kp, kk0 + 32 + lo, pitch, 0, hi);
      kf1 = load_frag(Kp, kk0 + 48 + lo, pitch, 0, hi);
    }
#pragma unroll
    for (int r = 0; r < 8; ++r) {
      float rm = half_max(fmaxf(s0[r], s1[r]) * SCALE);
      float mn = fmaxf(m[r], rm);
      float al = __expf(m[r] - mn);
      float p0 = __expf(s0[r] * SCALE - mn);
      float p1 = __expf(s1[r] * SCALE - mn);
      Z[r] = Z[r] * al + half_sum(p0 + p1);
      m[r] = mn;
    }
  }
  if (lo == 0) {
#pragma unroll
    for (int r = 0; r < 8; ++r) {
      int idx = (b * HN + h) * NQ + qt * 16 + r + 8 * hi;
      mrow[idx] = m[r];
      zrow[idx] = Z[r];
    }
  }
}

// ---------- guidance pass 2: text output + modulated image attention ----------
__global__ void k_guid_pass2(const uint16_t* __restrict__ Qm, const uint16_t* __restrict__ Km,
                             const uint16_t* __restrict__ Vm, const uint16_t* __restrict__ Qim,
                             const uint16_t* __restrict__ Kim, const uint16_t* __restrict__ Vim,
                             const float* __restrict__ mrow, const float* __restrict__ zrow,
                             uint16_t* __restrict__ Otm, uint16_t* __restrict__ Oim) {
  int lane = threadIdx.x, lo = lane & 15, hi = lane >> 4;
  int qt = blockIdx.x, b = blockIdx.y / HN, h = blockIdx.y % HN;
  const int pitch = BN * DM;
  const uint16_t* Qp = Qm + b * DM + h * HDn;
  const uint16_t* Kp = Km + b * DM + h * HDn;
  const uint16_t* Vp = Vm + b * DM + h * HDn;
  const uint16_t* Qip = Qim + b * DM + h * HDn;
  const uint16_t* Kip = Kim + b * DM + h * HDn;
  const uint16_t* Vip = Vim + b * DM + h * HDn;

  __shared__ __align__(16) uint16_t Pt[16 * 32];
  __shared__ __align__(16) uint16_t Pi[16 * 32];
  __shared__ __align__(16) uint16_t VTt[32 * 32];
  __shared__ __align__(16) uint16_t VTi[32 * 32];

  v16bf qf = load_frag(Qp, qt * 16 + lo, pitch, 0, hi);
  v16bf qif = load_frag(Qip, qt * 16 + lo, pitch, 0, hi);
  float mt[8], iZ[8], mi[8], Zi[8];
#pragma unroll
  for (int r = 0; r < 8; ++r) {
    int idx = (b * HN + h) * NQ + qt * 16 + r + 8 * hi;
    mt[r] = mrow[idx];
    iZ[r] = 1.0f / zrow[idx];
    mi[r] = -1e30f;
    Zi[r] = 0.0f;
  }
  v8f ot0 = v8f{}, ot1 = v8f{}, oi0 = v8f{}, oi1 = v8f{};

  for (int kk0 = 0; kk0 < NKK; kk0 += 32) {
    v16bf kf0 = load_frag(Kp, kk0 + lo, pitch, 0, hi);
    v16bf kf1 = load_frag(Kp, kk0 + 16 + lo, pitch, 0, hi);
    v16bf ki0 = load_frag(Kip, kk0 + lo, pitch, 0, hi);
    v16bf ki1 = load_frag(Kip, kk0 + 16 + lo, pitch, 0, hi);
    // stage both V chunks transposed; overlaps with the four S WMMAs below
    {
      const uint16_t* vr = Vp + (size_t)(kk0 + lane) * pitch;
      const uint16_t* vri = Vip + (size_t)(kk0 + lane) * pitch;
#pragma unroll
      for (int i = 0; i < 32; ++i) {
        VTt[i * 32 + lane] = vr[i];
        VTi[i * 32 + lane] = vri[i];
      }
    }
    v8f s0 = wmma_bf16(qf, kf0, v8f{});
    v8f s1 = wmma_bf16(qf, kf1, v8f{});
    v8f t0 = wmma_bf16(qif, ki0, v8f{});
    v8f t1 = wmma_bf16(qif, ki1, v8f{});
    float al[8];
#pragma unroll
    for (int r = 0; r < 8; ++r) {
      // exact text probabilities
      float p0 = __expf(s0[r] * SCALE - mt[r]) * iZ[r];
      float p1 = __expf(s1[r] * SCALE - mt[r]) * iZ[r];
      // modulated image logits
      float l0 = t0[r] * SCALE * p0;
      float l1 = t1[r] * SCALE * p1;
      float rm = half_max(fmaxf(l0, l1));
      float mn = fmaxf(mi[r], rm);
      al[r] = __expf(mi[r] - mn);
      float e0 = __expf(l0 - mn);
      float e1 = __expf(l1 - mn);
      Zi[r] = Zi[r] * al[r] + half_sum(e0 + e1);
      mi[r] = mn;
      Pt[(r + 8 * hi) * 32 + lo] = f2bf(p0);
      Pt[(r + 8 * hi) * 32 + 16 + lo] = f2bf(p1);
      Pi[(r + 8 * hi) * 32 + lo] = f2bf(e0);
      Pi[(r + 8 * hi) * 32 + 16 + lo] = f2bf(e1);
    }
    v16bf pft = load_frag(Pt, lo, 32, 0, hi);
    v16bf pfi = load_frag(Pi, lo, 32, 0, hi);
    v16bf vt0 = load_frag(VTt, lo, 32, 0, hi);
    v16bf vt1 = load_frag(VTt, 16 + lo, 32, 0, hi);
    v16bf vi0 = load_frag(VTi, lo, 32, 0, hi);
    v16bf vi1 = load_frag(VTi, 16 + lo, 32, 0, hi);
    ot0 = wmma_bf16(pft, vt0, ot0);
    ot1 = wmma_bf16(pft, vt1, ot1);
#pragma unroll
    for (int r = 0; r < 8; ++r) { oi0[r] *= al[r]; oi1[r] *= al[r]; }
    oi0 = wmma_bf16(pfi, vi0, oi0);
    oi1 = wmma_bf16(pfi, vi1, oi1);
  }
  uint16_t* Otp = Otm + b * DM + h * HDn;
  uint16_t* Oip = Oim + b * DM + h * HDn;
#pragma unroll
  for (int r = 0; r < 8; ++r) {
    size_t rowoff = (size_t)(qt * 16 + r + 8 * hi) * pitch;
    Otp[rowoff + lo] = f2bf(ot0[r]);          // text probs already normalized
    Otp[rowoff + 16 + lo] = f2bf(ot1[r]);
    float inv = 1.0f / Zi[r];
    Oip[rowoff + lo] = f2bf(oi0[r] * inv);
    Oip[rowoff + 16 + lo] = f2bf(oi1[r] * inv);
  }
}

// ---------- host orchestration ----------
extern "C" void kernel_launch(void* const* d_in, const int* in_sizes, int n_in,
                              void* d_out, int out_size, void* d_ws, size_t ws_size,
                              hipStream_t stream) {
  (void)in_sizes; (void)n_in; (void)out_size; (void)ws_size;
  auto F = [&](int i) { return (const float*)d_in[i]; };

  char* base = (char*)d_ws;
  size_t off = 0;
  auto alloc = [&](size_t bytes) -> void* {
    void* p = base + off;
    off += (bytes + 255) & ~(size_t)255;
    return p;
  };
  auto aU = [&](size_t n) { return (uint16_t*)alloc(n * 2); };
  auto aF = [&](size_t n) { return (float*)alloc(n * 4); };

  const size_t DD = (size_t)DM * DM, FD = (size_t)DFF * DM;
  // bf16 weights
  uint16_t *w_sa[4], *w_sai[4], *w_ga[8], *w_f1, *w_f2, *w_fi1, *w_fi2;
  for (int i = 0; i < 4; ++i) w_sa[i] = aU(DD);
  for (int i = 0; i < 4; ++i) w_sai[i] = aU(DD);
  for (int i = 0; i < 8; ++i) w_ga[i] = aU(DD);
  w_f1 = aU(FD); w_f2 = aU(FD); w_fi1 = aU(FD); w_fi2 = aU(FD);
  // activations
  const size_t AQ = (size_t)MQ * DM, AK = (size_t)MK * DM;
  uint16_t* qk_b = aU(AQ);   uint16_t* t2_b = aU(AQ);
  uint16_t* qb = aU(AQ);     uint16_t* kb = aU(AQ);      uint16_t* vb = aU(AQ);
  uint16_t* ao_b = aU(AQ);
  uint16_t* gq_in = aU(AQ);  uint16_t* gqi_in = aU(AQ);
  uint16_t* mem_pos = aU(AK); uint16_t* mem_b = aU(AK);  uint16_t* memim_b = aU(AK);
  uint16_t* gq_b = aU(AQ);   uint16_t* gqi_b = aU(AQ);
  uint16_t* gk_b = aU(AK);   uint16_t* gv_b = aU(AK);
  uint16_t* gki_b = aU(AK);  uint16_t* gvi_b = aU(AK);
  uint16_t* got_b = aU(AQ);  uint16_t* goi_b = aU(AQ);
  uint16_t* h_b = aU(AQ);    uint16_t* h1_b = aU((size_t)MQ * DFF);
  float* tgt1 = aF(AQ);  float* timg1 = aF(AQ);
  float* tgt2 = aF(AQ);  float* timg2 = aF(AQ);
  float* mrow = aF((size_t)BN * HN * NQ);
  float* zrow = aF((size_t)BN * HN * NQ);
  float* outT = (float*)d_out;
  float* outI = outT + AQ;

  auto cvt = [&](int idx, uint16_t* dst, size_t n) {
    k_f2bf<<<dim3((unsigned)((n + 255) / 256)), 256, 0, stream>>>(F(idx), dst, (int)n);
  };
  auto gemm = [&](const uint16_t* A, const uint16_t* W, const float* bias,
                  const float* resid, float* outf, uint16_t* outb,
                  int M, int N, int K, int relu) {
    k_gemm<<<dim3(M / 64, N / 64), 128, 0, stream>>>(A, W, bias, resid, outf, outb, M, N, K, relu);
  };
  auto ln = [&](const float* x, int gi, int bi, const float* pos, uint16_t* op,
                uint16_t* opl) {
    k_ln<<<dim3(MQ), DM, 0, stream>>>(x, F(gi), F(bi), pos, op, opl);
  };

  // ---- weight conversion ----
  for (int i = 0; i < 4; ++i) cvt(6 + i, w_sa[i], DD);    // sa_wq/wk/wv/wo
  for (int i = 0; i < 4; ++i) cvt(14 + i, w_sai[i], DD);  // sai_*
  for (int i = 0; i < 8; ++i) cvt(22 + i, w_ga[i], DD);   // ga_wq,wqi,wk,wki,wv,wvi,wp,wpi
  cvt(50, w_f1, FD); cvt(52, w_f2, FD); cvt(54, w_fi1, FD); cvt(56, w_fi2, FD);

  // ---- self-attention (text) ----
  ln(F(0), 38, 39, F(5), qk_b, t2_b);                       // ln1 (+query_pos)
  gemm(qk_b, w_sa[0], F(10), nullptr, nullptr, qb, MQ, DM, DM, 0);
  gemm(qk_b, w_sa[1], F(11), nullptr, nullptr, kb, MQ, DM, DM, 0);
  gemm(t2_b, w_sa[2], F(12), nullptr, nullptr, vb, MQ, DM, DM, 0);
  k_flash_self<<<dim3(NQ / 16, BN * HN), 32, 0, stream>>>(qb, kb, vb, ao_b, NQ);
  gemm(ao_b, w_sa[3], F(13), F(0), tgt1, nullptr, MQ, DM, DM, 0);

  // ---- self-attention (image) ----
  ln(F(1), 40, 41, nullptr, nullptr, t2_b);                 // ln1i, q=k=v
  gemm(t2_b, w_sai[0], F(18), nullptr, nullptr, qb, MQ, DM, DM, 0);
  gemm(t2_b, w_sai[1], F(19), nullptr, nullptr, kb, MQ, DM, DM, 0);
  gemm(t2_b, w_sai[2], F(20), nullptr, nullptr, vb, MQ, DM, DM, 0);
  k_flash_self<<<dim3(NQ / 16, BN * HN), 32, 0, stream>>>(qb, kb, vb, ao_b, NQ);
  gemm(ao_b, w_sai[3], F(21), F(1), timg1, nullptr, MQ, DM, DM, 0);

  // ---- guidance attention ----
  ln(tgt1, 42, 43, F(5), gq_in, nullptr);                   // ln2 (+query_pos)
  ln(timg1, 44, 45, nullptr, nullptr, gqi_in);              // ln2i
  {
    int n = (int)AK;
    k_add_f2bf<<<dim3((n + 255) / 256), 256, 0, stream>>>(F(2), F(4), mem_pos, n);
    k_add_f2bf<<<dim3((n + 255) / 256), 256, 0, stream>>>(F(2), nullptr, mem_b, n);
    k_add_f2bf<<<dim3((n + 255) / 256), 256, 0, stream>>>(F(3), nullptr, memim_b, n);
  }
  gemm(gq_in, w_ga[0], F(30), nullptr, nullptr, gq_b, MQ, DM, DM, 0);    // ga_wq
  gemm(gqi_in, w_ga[1], F(31), nullptr, nullptr, gqi_b, MQ, DM, DM, 0);  // ga_wqi
  gemm(mem_pos, w_ga[2], F(32), nullptr, nullptr, gk_b, MK, DM, DM, 0);  // ga_wk
  gemm(memim_b, w_ga[3], F(33), nullptr, nullptr, gki_b, MK, DM, DM, 0); // ga_wki
  gemm(mem_b, w_ga[4], F(34), nullptr, nullptr, gv_b, MK, DM, DM, 0);    // ga_wv
  gemm(memim_b, w_ga[5], F(35), nullptr, nullptr, gvi_b, MK, DM, DM, 0); // ga_wvi
  k_guid_pass1<<<dim3(NQ / 16, BN * HN), 32, 0, stream>>>(gq_b, gk_b, mrow, zrow);
  k_guid_pass2<<<dim3(NQ / 16, BN * HN), 32, 0, stream>>>(gq_b, gk_b, gv_b, gqi_b,
                                                          gki_b, gvi_b, mrow, zrow,
                                                          got_b, goi_b);
  gemm(got_b, w_ga[6], F(36), tgt1, tgt2, nullptr, MQ, DM, DM, 0);   // ga_wp
  gemm(goi_b, w_ga[7], F(37), timg1, timg2, nullptr, MQ, DM, DM, 0); // ga_wpi

  // ---- FFN (text) ----
  ln(tgt2, 46, 47, nullptr, nullptr, h_b);                  // ln3
  gemm(h_b, w_f1, F(51), nullptr, nullptr, h1_b, MQ, DFF, DM, 1);
  gemm(h1_b, w_f2, F(53), tgt2, outT, nullptr, MQ, DM, DFF, 0);

  // ---- FFN (image) ----
  ln(timg2, 48, 49, nullptr, nullptr, h_b);                 // ln3i
  gemm(h_b, w_fi1, F(55), nullptr, nullptr, h1_b, MQ, DFF, DM, 1);
  gemm(h1_b, w_fi2, F(57), timg2, outI, nullptr, MQ, DM, DFF, 0);
}
